// Encoder_34840774705561
// MI455X (gfx1250) — matrix operand order repaired and checked by reading, not compiled, not run on hardware
//
#include <hip/hip_runtime.h>
#include <math.h>

// ---------------------------------------------------------------------------
// Bidirectional LSTM encoder for MI455X (gfx1250), wave32 + WMMA bf16 + TDM.
//   VOCAB=32000, EMB=512, HID=1024, SEQ=512, BATCH=64
// Pipeline:
//   1) init        : zero h double-buffers + grid-barrier words
//   2) embed       : gather emb_w rows (row 0 forced to 0) -> bf16 [32768][512]
//   3) swizzle x4  : Wih/Whh fp32 -> bf16 in exact WMMA B-matrix lane layout
//   4) xproj  x2   : X @ Wih^T + (bih+bhh) -> fp32 [T*B][4096]; A tile staged
//                    to LDS via TDM, B software-pipelined       (WMMA bf16)
//   5) lstm        : persistent kernel, 32 WGs/direction, grid barrier per
//                    step; h_prev staged into LDS via TENSOR_LOAD_TO_LDS with
//                    bank-skew padding; per-WG 64x128x1024 WMMA GEMM with A
//                    from LDS, B pipelined; x_proj loads hoisted; c in LDS;
//                    h double-buffered in global (L2)
//   6) add         : d_out += hs_b
// ---------------------------------------------------------------------------

typedef __bf16 bf16_t;
typedef __attribute__((ext_vector_type(16))) __bf16 v16bf;
typedef __attribute__((ext_vector_type(8)))  __bf16 v8bf;
typedef __attribute__((ext_vector_type(8)))  float  v8f;
typedef __attribute__((ext_vector_type(4)))  unsigned int u32x4;
typedef __attribute__((ext_vector_type(8)))  int i32x8;
typedef __attribute__((ext_vector_type(4)))  int i32x4;

#define T_SEQ  512
#define BATCH  64
#define EMB    512
#define HID    1024
#define G4     4096
#define NSLICE 32     // workgroups per direction
#define HTILE  32     // cells per workgroup
#define NLOC   128    // 4*HTILE gate columns per workgroup
#define RSTRIDE  1032 // lstm LDS h row stride, halfs (2064 B: 4-bank skew)
#define XRSTRIDE 520  // xproj LDS X row stride, halfs (1040 B: 4-bank skew)

// ---- workspace layout (bytes) ---------------------------------------------
#define OFF_XPF   0ull
#define SZ_XP     (512ull*64*4096*4)            // 512 MB
#define OFF_XPB   (OFF_XPF + SZ_XP)
#define OFF_HSB   (OFF_XPB + SZ_XP)
#define SZ_HSB    (64ull*512*1024*4)            // 128 MB
#define OFF_EMB   (OFF_HSB + SZ_HSB)
#define SZ_EMB    (32768ull*512*2)              // 32 MB
#define OFF_WIHSF (OFF_EMB + SZ_EMB)
#define SZ_WIHS   (512ull*4096*2)               // 4 MB
#define OFF_WIHSB (OFF_WIHSF + SZ_WIHS)
#define OFF_WHHSF (OFF_WIHSB + SZ_WIHS)
#define SZ_WHHS   (1024ull*4096*2)              // 8 MB
#define OFF_WHHSB (OFF_WHHSF + SZ_WHHS)
#define OFF_HBUF  (OFF_WHHSB + SZ_WHHS)
#define SZ_HBUF1  (64ull*1024*2)                // one h buffer (bf16)
#define OFF_BAR   (OFF_HBUF + 4*SZ_HBUF1)

// ---------------------------------------------------------------------------
__global__ __launch_bounds__(256) void init_kernel(void* ws) {
  unsigned* hb = (unsigned*)((char*)ws + OFF_HBUF);
  unsigned i = blockIdx.x * 256u + threadIdx.x;
  if (i < (4u * (unsigned)SZ_HBUF1) / 4u) hb[i] = 0u;
  if (i < 8u) ((unsigned*)((char*)ws + OFF_BAR))[i] = 0u;
}

// ---------------------------------------------------------------------------
__global__ __launch_bounds__(256) void embed_kernel(const int* __restrict__ src,
                                                    const float* __restrict__ emb_w,
                                                    bf16_t* __restrict__ out) {
  unsigned idx = blockIdx.x * 256u + threadIdx.x;   // over 32768*512
  int tok = idx >> 9;
  int e   = idx & 511;
  int row = src[tok];
  float v = (row == 0) ? 0.f : emb_w[(size_t)row * EMB + e];
  out[idx] = (bf16_t)v;
}

// ---------------------------------------------------------------------------
// Pack a row-major weight W[N][K] (fp32) as B-matrix tiles (32Kx16N, bf16)
// in per-lane order: halfs[((nt*nkt + kt)*32 + lane)*16 + e] where
//   k = kt*32 + 8*(lane/16) + (e&7) + 16*(e/8) ,  n = nt*16 + lane%16.
// mode 0: Wih (K=512), natural column order n.
// mode 1: Whh (K=1024), grouped per WG slice: [slice][nt(8)][kt(32)][lane][e],
//   n_local = nt*16+lane%16 -> row = (n_local/32)*1024 + slice*32 + n_local%32
__global__ __launch_bounds__(256) void swizzle_kernel(const float* __restrict__ W,
                                                      bf16_t* __restrict__ out,
                                                      int mode) {
  unsigned idx = blockIdx.x * 256u + threadIdx.x;
  int e    = idx & 15;
  int lane = (idx >> 4) & 31;
  int k, row, K;
  if (mode == 0) {                      // Wih: 2,097,152 halfs
    int kt = (idx >> 9) & 15;
    int nt = idx >> 13;
    K = 512;
    k = kt * 32 + ((lane >> 4) << 3) + (e & 7) + ((e >> 3) << 4);
    row = nt * 16 + (lane & 15);
  } else {                              // Whh: 4,194,304 halfs
    int kt    = (idx >> 9) & 31;
    int nt    = (idx >> 14) & 7;
    int slice = idx >> 17;
    K = 1024;
    k = kt * 32 + ((lane >> 4) << 3) + (e & 7) + ((e >> 3) << 4);
    int nl   = nt * 16 + (lane & 15);
    row = (nl >> 5) * 1024 + slice * HTILE + (nl & 31);
  }
  out[idx] = (bf16_t)W[(size_t)row * K + k];
}

// ---------------------------------------------------------------------------
// TDM: DMA a ROWS x DIM0 bf16 tile (row-major, stride DIM0) from global into
// LDS, inserting (PAD_AMT+1) DWORDs of padding after every 256 DWORDs (1 KB)
// so LDS rows get a bank skew. Descriptor per CDNA5 ISA sec 8.3/8.4.
// Issued by one wave; EXEC ignored by TDM.
template <int DIM0, int ROWS, int PAD_AMT>
__device__ __forceinline__ void tdm_load_2d(unsigned lds_off,
                                            const bf16_t* gsrc) {
  unsigned long long ga = (unsigned long long)(size_t)gsrc;
  const unsigned long long s1 = (unsigned long long)DIM0 * ROWS;
  u32x4 g0 = { 1u,                       // count=1, user descriptor
               lds_off,                  // lds_addr (bytes)
               (unsigned)(ga & 0xffffffffull),
               (unsigned)(ga >> 32) | (2u << 30) };   // addr[56:32] | type=2
  i32x8 g1 = { (int)((1u << 16)          // data_size = 2 bytes
                   | (1u << 20)          // pad_enable
                   | (7u << 22)          // pad_interval: every 256 DWORDs
                   | ((unsigned)PAD_AMT << 25)),      // pad_amount code
               (int)((unsigned)DIM0 << 16),           // tensor_dim0 lo16
               (int)((unsigned)ROWS << 16),           // dim0 hi=0 | tensor_dim1
               (int)((unsigned)DIM0 << 16),           // dim1 hi=0 | tile_dim0
               ROWS,                                  // tile_dim1 | tile_dim2=0
               DIM0,                                  // tensor_dim0_stride
               (int)((s1 & 0xffffull) << 16),         // stride0 hi | stride1 lo
               (int)(s1 >> 16) };                     // stride1 hi
  i32x4 gz4 = { 0, 0, 0, 0 };
  i32x8 gz8 = { 0, 0, 0, 0, 0, 0, 0, 0 };
  __builtin_amdgcn_tensor_load_to_lds(g0, g1, gz4, gz4, gz8, 0);
  __builtin_amdgcn_s_wait_tensorcnt(0);
}

// ---------------------------------------------------------------------------
// x_proj = X(bf16 [32768][512]) @ Wih^T + (bih+bhh) -> fp32 [32768][4096]
// A tile (64x512, 64 KB) staged into LDS by TDM; B software-pipelined.
__global__ __launch_bounds__(256) void xproj_kernel(const bf16_t* __restrict__ X,
                                                    const bf16_t* __restrict__ WS,
                                                    const float* __restrict__ bih,
                                                    const float* __restrict__ bhh,
                                                    float* __restrict__ out) {
  const int m0   = blockIdx.x * 64;
  const int n0   = blockIdx.y * 128;
  const int wv   = threadIdx.x >> 5;
  const int lane = threadIdx.x & 31;
  const int ntg  = (n0 >> 4) + wv;

  __shared__ __align__(16) unsigned char xsm[64 * XRSTRIDE * 2];  // 66,560 B
  bf16_t* xstage = (bf16_t*)xsm;

  if (threadIdx.x < 32)   // pad 16 B per 1024 B row -> 4-bank skew
    tdm_load_2d<EMB, 64, 3>((unsigned)(size_t)xsm, X + (size_t)m0 * EMB);
  __syncthreads();

  v8f acc[4] = {};
  v16bf bcur = *(const v16bf*)(WS + ((size_t)(ntg * 16) * 32 + lane) * 16);
  for (int kt = 0; kt < 16; ++kt) {
    v16bf bnx = bcur;
    if (kt < 15)
      bnx = *(const v16bf*)(WS + ((size_t)(ntg * 16 + kt + 1) * 32 + lane) * 16);
#pragma unroll
    for (int mt = 0; mt < 4; ++mt) {
      const bf16_t* ar = xstage + (mt * 16 + (lane & 15)) * XRSTRIDE
                                + kt * 32 + ((lane >> 4) << 3);
      v8bf lo = *(const v8bf*)ar;
      v8bf hi = *(const v8bf*)(ar + 16);
      v16bf a = __builtin_shufflevector(lo, hi, 0, 1, 2, 3, 4, 5, 6, 7,
                                        8, 9, 10, 11, 12, 13, 14, 15);
      acc[mt] = __builtin_amdgcn_wmma_f32_16x16x32_bf16(
          false, a, false, bcur, (short)0, acc[mt], false, false);
    }
    bcur = bnx;
  }
  const int col = n0 + wv * 16 + (lane & 15);
  const float bias = bih[col] + bhh[col];
#pragma unroll
  for (int mt = 0; mt < 4; ++mt)
#pragma unroll
    for (int r = 0; r < 8; ++r) {
      int m = m0 + mt * 16 + ((lane >> 4) << 3) + r;
      out[(size_t)m * G4 + col] = acc[mt][r] + bias;
    }
}

// ---------------------------------------------------------------------------
__device__ __forceinline__ void dir_barrier(unsigned* cnt, unsigned* gen,
                                            unsigned nwg) {
  __threadfence();
  __syncthreads();
  if (threadIdx.x == 0) {
    unsigned g = __hip_atomic_load(gen, __ATOMIC_ACQUIRE,
                                   __HIP_MEMORY_SCOPE_AGENT);
    unsigned a = __hip_atomic_fetch_add(cnt, 1u, __ATOMIC_ACQ_REL,
                                        __HIP_MEMORY_SCOPE_AGENT);
    if (a == nwg - 1u) {
      __hip_atomic_store(cnt, 0u, __ATOMIC_RELEASE, __HIP_MEMORY_SCOPE_AGENT);
      __hip_atomic_fetch_add(gen, 1u, __ATOMIC_ACQ_REL,
                             __HIP_MEMORY_SCOPE_AGENT);
    } else {
      while (__hip_atomic_load(gen, __ATOMIC_ACQUIRE,
                               __HIP_MEMORY_SCOPE_AGENT) == g)
        __builtin_amdgcn_s_sleep(2);
    }
  }
  __syncthreads();
  __threadfence();
}

// Persistent recurrent kernel: 64 WGs (32 fwd + 32 bwd), 256 threads each.
__global__ __launch_bounds__(256) void lstm_kernel(
    const float* __restrict__ xpF, const float* __restrict__ xpB,
    const bf16_t* __restrict__ whhF, const bf16_t* __restrict__ whhB,
    bf16_t* __restrict__ hbuf, unsigned* __restrict__ bar,
    float* __restrict__ outF /*d_out [B][T][H]*/,
    float* __restrict__ outB /*hs_b  [B][T][H]*/) {
  const int wg    = blockIdx.x;
  const int dir   = wg >> 5;
  const int slice = wg & 31;
  const int j0    = slice * HTILE;

  const float*  xp   = dir ? xpB : xpF;
  const bf16_t* whh  = (dir ? whhB : whhF) + (size_t)slice * (NLOC * HID);
  float*        outp = dir ? outB : outF;
  bf16_t* hb0 = hbuf + (size_t)(dir * 2 + 0) * (BATCH * HID);
  bf16_t* hb1 = hbuf + (size_t)(dir * 2 + 1) * (BATCH * HID);
  unsigned* cnt = bar + dir * 2;
  unsigned* gen = bar + dir * 2 + 1;

  // h-stage (64 rows x 2064 B skewed) ALIASED with gates[64][128] f32:
  // h-stage is live only during the GEMM, gates only after it.
  __shared__ __align__(16) unsigned char smem[BATCH * RSTRIDE * 2]; // 132,096 B
  __shared__ float cst[BATCH][HTILE];                               //   8 KB
  bf16_t* hstage = (bf16_t*)smem;
  float*  gates  = (float*)smem;                                    // [64][128]
  const unsigned lds_off = (unsigned)(size_t)smem;   // LDS byte offset

  const int wv   = threadIdx.x >> 5;
  const int lane = threadIdx.x & 31;

  for (int i = threadIdx.x; i < BATCH * HTILE; i += 256)
    ((float*)cst)[i] = 0.f;
  __syncthreads();

  for (int s = 0; s < T_SEQ; ++s) {
    const int t = dir ? (T_SEQ - 1 - s) : s;
    const bf16_t* hprev = (s & 1) ? hb1 : hb0;
    bf16_t*       hnext = (s & 1) ? hb0 : hb1;

    // ---- hoist this step's x_proj gate loads (HBM) above the GEMM ---------
    float xv[8][4];
#pragma unroll
    for (int q = 0; q < 8; ++q) {
      const int cell = threadIdx.x + 256 * q;
      const int m  = cell >> 5;
      const int jc = cell & 31;
      const float* xr = xp + (size_t)(t * BATCH + m) * G4 + j0 + jc;
      xv[q][0] = xr[0];
      xv[q][1] = xr[1024];
      xv[q][2] = xr[2048];
      xv[q][3] = xr[3072];
    }

    // ---- stage h_prev into LDS via Tensor Data Mover (wave 0 only) --------
    if (threadIdx.x < 32) {   // pad 2x8 B per 2048 B row -> 4-bank skew
      tdm_load_2d<HID, BATCH, 1>(lds_off, hprev);
    }
    __syncthreads();

    // ---- GEMM: gates[64][128] = h[64][1024] @ Whh_slice^T (A from LDS) ----
    v8f acc[4] = {};
    v16bf bcur = *(const v16bf*)(whh + ((size_t)(wv * 32) * 32 + lane) * 16);
    for (int kt = 0; kt < 32; ++kt) {
      v16bf bnx = bcur;
      if (kt < 31)
        bnx = *(const v16bf*)(whh +
                              ((size_t)(wv * 32 + kt + 1) * 32 + lane) * 16);
#pragma unroll
      for (int mt = 0; mt < 4; ++mt) {
        const bf16_t* ar = hstage + (mt * 16 + (lane & 15)) * RSTRIDE
                                  + kt * 32 + ((lane >> 4) << 3);
        v8bf lo = *(const v8bf*)ar;
        v8bf hi = *(const v8bf*)(ar + 16);
        v16bf a = __builtin_shufflevector(lo, hi, 0, 1, 2, 3, 4, 5, 6, 7,
                                          8, 9, 10, 11, 12, 13, 14, 15);
        acc[mt] = __builtin_amdgcn_wmma_f32_16x16x32_bf16(
            false, a, false, bcur, (short)0, acc[mt], false, false);
      }
      bcur = bnx;
    }
    __syncthreads();   // all h-stage reads done before gates overwrite (alias)

#pragma unroll
    for (int mt = 0; mt < 4; ++mt)
#pragma unroll
      for (int r = 0; r < 8; ++r)
        gates[(mt * 16 + ((lane >> 4) << 3) + r) * NLOC + wv * 16 + (lane & 15)]
            = acc[mt][r];
    __syncthreads();

    // ---- elementwise gate math, 2048 cells over 256 threads ---------------
#pragma unroll
    for (int q = 0; q < 8; ++q) {
      const int cell = threadIdx.x + 256 * q;
      const int m  = cell >> 5;
      const int jc = cell & 31;
      float gi = xv[q][0] + gates[m * NLOC + jc];
      float gf = xv[q][1] + gates[m * NLOC + 32 + jc];
      float gg = xv[q][2] + gates[m * NLOC + 64 + jc];
      float go = xv[q][3] + gates[m * NLOC + 96 + jc];
      float ii = 1.f / (1.f + __expf(-gi));
      float ff = 1.f / (1.f + __expf(-gf));
      float g  = tanhf(gg);
      float oo = 1.f / (1.f + __expf(-go));
      float c  = ff * cst[m][jc] + ii * g;
      cst[m][jc] = c;
      float h = oo * tanhf(c);
      hnext[(size_t)m * HID + j0 + jc] = (bf16_t)h;
      outp[((size_t)m * T_SEQ + t) * HID + j0 + jc] = h;
    }
    dir_barrier(cnt, gen, NSLICE);  // also orders gates reads vs next TDM
  }
}

// ---------------------------------------------------------------------------
__global__ __launch_bounds__(256) void add_kernel(float* __restrict__ out,
                                                  const float* __restrict__ hsb) {
  size_t i = ((size_t)blockIdx.x * 256 + threadIdx.x) * 4;
  float4 a = *(const float4*)(out + i);
  float4 b = *(const float4*)(hsb + i);
  a.x += b.x; a.y += b.y; a.z += b.z; a.w += b.w;
  *(float4*)(out + i) = a;
}

// ---------------------------------------------------------------------------
extern "C" void kernel_launch(void* const* d_in, const int* in_sizes, int n_in,
                              void* d_out, int out_size, void* d_ws,
                              size_t ws_size, hipStream_t stream) {
  (void)in_sizes; (void)n_in; (void)out_size; (void)ws_size;
  const int*   src   = (const int*)d_in[0];
  const float* emb_w = (const float*)d_in[1];
  const float* WihF  = (const float*)d_in[2];
  const float* WhhF  = (const float*)d_in[3];
  const float* bihF  = (const float*)d_in[4];
  const float* bhhF  = (const float*)d_in[5];
  const float* WihB  = (const float*)d_in[6];
  const float* WhhB  = (const float*)d_in[7];
  const float* bihB  = (const float*)d_in[8];
  const float* bhhB  = (const float*)d_in[9];

  char* ws = (char*)d_ws;
  float*   xpF   = (float*)(ws + OFF_XPF);
  float*   xpB   = (float*)(ws + OFF_XPB);
  float*   hsb   = (float*)(ws + OFF_HSB);
  bf16_t*  emb   = (bf16_t*)(ws + OFF_EMB);
  bf16_t*  wihsF = (bf16_t*)(ws + OFF_WIHSF);
  bf16_t*  wihsB = (bf16_t*)(ws + OFF_WIHSB);
  bf16_t*  whhsF = (bf16_t*)(ws + OFF_WHHSF);
  bf16_t*  whhsB = (bf16_t*)(ws + OFF_WHHSB);
  bf16_t*  hbuf  = (bf16_t*)(ws + OFF_HBUF);
  unsigned* bar  = (unsigned*)(ws + OFF_BAR);

  init_kernel<<<512, 256, 0, stream>>>(d_ws);
  embed_kernel<<<65536, 256, 0, stream>>>(src, emb_w, emb);
  swizzle_kernel<<<8192,  256, 0, stream>>>(WihF, wihsF, 0);
  swizzle_kernel<<<8192,  256, 0, stream>>>(WihB, wihsB, 0);
  swizzle_kernel<<<16384, 256, 0, stream>>>(WhhF, whhsF, 1);
  swizzle_kernel<<<16384, 256, 0, stream>>>(WhhB, whhsB, 1);
  dim3 gg(512, 32);
  xproj_kernel<<<gg, 256, 0, stream>>>(emb, wihsF, bihF, bhhF, xpF);
  xproj_kernel<<<gg, 256, 0, stream>>>(emb, wihsB, bihB, bhhB, xpB);
  lstm_kernel<<<64, 256, 0, stream>>>(xpF, xpB, whhsF, whhsB, hbuf, bar,
                                      (float*)d_out, hsb);
  add_kernel<<<32768, 256, 0, stream>>>((float*)d_out, hsb);
}
